// LinearAttention_52707838656610
// MI455X (gfx1250) — matrix operand order repaired and checked by reading, not compiled, not run on hardware
//
#include <hip/hip_runtime.h>
#include <hip/hip_bf16.h>

// ---------------------------------------------------------------------------
// Causal linear attention (ELU+1 feature map) for MI455X / gfx1250.
// B=2, T=2048, D=512, H=8, dh=dv=64.  All GEMMs via v_wmma_f32_16x16x32_f16
// (f16 operands, f32 accumulation).  Causal scan parallelized with 64-wide
// chunks: per-chunk KV GEMMs -> exclusive chunk scan -> per-chunk attention.
// ---------------------------------------------------------------------------

typedef __attribute__((ext_vector_type(16))) _Float16 v16h;
typedef __attribute__((ext_vector_type(8)))  float    v8f;

#define BQ 2
#define TT 2048
#define DM 512
#define HH 8
#define DH 64
#define NC 32     // chunks of 64 along T
#define CH 64     // chunk length

__device__ __forceinline__ v8f wmma16(v16h a, v16h b, v8f c) {
  // v_wmma_f32_16x16x32_f16: (neg_a, A, neg_b, B, c_mod, C, reuse_a, reuse_b)
  return __builtin_amdgcn_wmma_f32_16x16x32_f16(false, a, false, b, (short)0, c,
                                                false, false);
}

// 16-bit A-matrix 16x32 lane layout (ISA 7.12.2):
// lane l: m = l&15 ; K = (v>=4?16:0) + (v&3)*2 + e + ((l>=16)?8:0)
__device__ __forceinline__ int a_k(int idx, int lane) {
  int v = idx >> 1, e = idx & 1;
  return ((v & 4) ? 16 : 0) + ((v & 3) << 1) + e + ((lane >> 4) << 3);
}
// 16-bit B-matrix 32x16 lane layout: lane l: n = l&15 ; K = (v>=4?16:0) + (v&3)*2 + e + ((l>=16)?8:0)
__device__ __forceinline__ int b_k(int idx, int lane) {
  int v = idx >> 1, e = idx & 1;
  return ((v & 4) ? 16 : 0) + ((v & 3) << 1) + e + ((lane >> 4) << 3);
}
// f32 C/D 16x16: vgpr v, lane l -> m = v + ((l>=16)?8:0), n = l&15

__device__ __forceinline__ float phi(float x) {  // elu(x)+1
  return x > 0.f ? x + 1.f : __expf(x);
}

// ---------------------------------------------------------------------------
// Kernel 1: fused Q/K/V projection.  y = x @ W^T  (W is [out,in] row-major).
// grid = (8 o-tiles, 64 n-tiles, 3 projections), block = 256 (8 waves).
// Each block produces a 64x64 output tile; each wave owns two 16x16 tiles.
// phi() applied for Q,K; result stored f16 in [B,H,T,dh] layout.
// ---------------------------------------------------------------------------
__global__ void proj_qkv_kernel(const float* __restrict__ x,
                                const float* __restrict__ Wq,
                                const float* __restrict__ Wk,
                                const float* __restrict__ Wv,
                                _Float16* __restrict__ Qh,
                                _Float16* __restrict__ Kh,
                                _Float16* __restrict__ Vh) {
  __shared__ __align__(16) _Float16 Xs[64 * 64];
  __shared__ __align__(16) _Float16 Wsh[64 * 64];

  const int o0 = blockIdx.x * 64;
  const int n0 = blockIdx.y * 64;
  const int p  = blockIdx.z;           // 0=Q, 1=K, 2=V
  const float* W = (p == 0) ? Wq : (p == 1) ? Wk : Wv;
  _Float16* dst  = (p == 0) ? Qh : (p == 1) ? Kh : Vh;

  const int t    = threadIdx.x;
  const int wave = t >> 5;
  const int lane = t & 31;
  const int ml   = lane & 15;                 // m within 16x16 tile (A rows)
  const int nl   = lane & 15;                 // n within tile (B cols)
  const int hi8  = (lane >> 4) << 3;          // C/D row offset

  v8f acc[2] = {v8f{}, v8f{}};

  for (int k0 = 0; k0 < DM; k0 += 64) {
    // stage x[n0:+64, k0:+64] and W[o0:+64, k0:+64] as f16
    for (int r = 0; r < 4; ++r) {
      int idx = t + r * 256;                  // 0..1023
      int row = idx >> 4;
      int c4  = (idx & 15) << 2;
      float4 xv = *(const float4*)&x[(n0 + row) * DM + k0 + c4];
      Xs[row * 64 + c4 + 0] = (_Float16)xv.x;
      Xs[row * 64 + c4 + 1] = (_Float16)xv.y;
      Xs[row * 64 + c4 + 2] = (_Float16)xv.z;
      Xs[row * 64 + c4 + 3] = (_Float16)xv.w;
      float4 wv = *(const float4*)&W[(o0 + row) * DM + k0 + c4];
      Wsh[row * 64 + c4 + 0] = (_Float16)wv.x;
      Wsh[row * 64 + c4 + 1] = (_Float16)wv.y;
      Wsh[row * 64 + c4 + 2] = (_Float16)wv.z;
      Wsh[row * 64 + c4 + 3] = (_Float16)wv.w;
    }
    __syncthreads();

    for (int tt = 0; tt < 2; ++tt) {
      int tile = wave * 2 + tt;
      int tm = tile >> 2, tn = tile & 3;
      for (int ks = 0; ks < 64; ks += 32) {
        v16h a, b;
        for (int i = 0; i < 16; ++i) {
          a[i] = Xs[(tm * 16 + ml) * 64 + ks + a_k(i, lane)];       // A[m][k]=x
          b[i] = Wsh[(tn * 16 + nl) * 64 + ks + b_k(i, lane)];      // B[k][n]=W[o][k]
        }
        acc[tt] = wmma16(a, b, acc[tt]);
      }
    }
    __syncthreads();
  }

  // epilogue: phi for Q/K, scatter to [B,H,T,dh]
  for (int tt = 0; tt < 2; ++tt) {
    int tile = wave * 2 + tt;
    int tm = tile >> 2, tn = tile & 3;
    for (int v = 0; v < 8; ++v) {
      int n = n0 + tm * 16 + v + hi8;         // token index (b*T + t)
      int o = o0 + tn * 16 + nl;              // h*64 + d
      float val = acc[tt][v];
      if (p < 2) val = phi(val);
      int b = n >> 11, tok = n & (TT - 1);
      int h = o >> 6,  d   = o & 63;
      dst[((b * HH + h) * TT + tok) * DH + d] = (_Float16)val;
    }
  }
}

// ---------------------------------------------------------------------------
// Kernel 2: per-chunk KV = K_c^T @ V_c  ([dh x dv]) and z_c = sum_i K_i.
// grid = (NC chunks, B*H), block = 256.
// ---------------------------------------------------------------------------
__global__ void chunk_kv_kernel(const _Float16* __restrict__ Kh,
                                const _Float16* __restrict__ Vh,
                                float* __restrict__ KVw,
                                float* __restrict__ zw) {
  __shared__ __align__(16) _Float16 Ks[64 * 64];
  __shared__ __align__(16) _Float16 Vs[64 * 64];

  const int c  = blockIdx.x;
  const int bh = blockIdx.y;
  const int t  = threadIdx.x;
  const int wave = t >> 5, lane = t & 31;
  const int ml = lane & 15, nl = lane & 15, hi8 = (lane >> 4) << 3;

  const long base = (long)(bh * TT + c * CH) * DH;   // halves
  for (int r = 0; r < 2; ++r) {
    int idx = t + r * 256;                           // 512 x uint4 = 4096 halves
    ((uint4*)Ks)[idx] = ((const uint4*)(Kh + base))[idx];
    ((uint4*)Vs)[idx] = ((const uint4*)(Vh + base))[idx];
  }
  __syncthreads();

  for (int tt = 0; tt < 2; ++tt) {
    int tile = wave * 2 + tt;
    int tm = tile >> 2, tn = tile & 3;               // tm over dh, tn over dv
    v8f acc = v8f{};
    for (int ks = 0; ks < 64; ks += 32) {            // contract over chunk rows i
      v16h a, b;
      for (int i = 0; i < 16; ++i) {
        // A[m=d][k=i] = K[i][d]  (transposed gather from LDS)
        a[i] = Ks[(ks + a_k(i, lane)) * 64 + tm * 16 + ml];
        // B[k=i][n=j] = V[i][j]
        b[i] = Vs[(ks + b_k(i, lane)) * 64 + tn * 16 + nl];
      }
      acc = wmma16(a, b, acc);
    }
    float* out = KVw + (long)(bh * NC + c) * (DH * DH);
    for (int v = 0; v < 8; ++v) {
      int m = tm * 16 + v + hi8;                     // d
      int n = tn * 16 + nl;                          // j
      out[m * DH + n] = acc[v];
    }
  }

  if (t < DH) {
    float s = 0.f;
    for (int i = 0; i < CH; ++i) s += (float)Ks[i * 64 + t];
    zw[(bh * NC + c) * DH + t] = s;
  }
}

// ---------------------------------------------------------------------------
// Kernel 3: in-place exclusive scan of KV_c / z_c over chunks, per (b,h).
// grid = B*H blocks, block = 256 (each thread owns 16 of the 4096 S elems).
// ---------------------------------------------------------------------------
__global__ void chunk_scan_kernel(float* __restrict__ KVw,
                                  float* __restrict__ zw) {
  const int bh = blockIdx.x;
  const int t  = threadIdx.x;
  float s[16];
  for (int i = 0; i < 16; ++i) s[i] = 0.f;
  float sz = 0.f;

  for (int c = 0; c < NC; ++c) {
    float* base = KVw + (long)(bh * NC + c) * (DH * DH);
    for (int i = 0; i < 16; ++i) {
      int e = i * 256 + t;
      float tmp = base[e];
      base[e] = s[i];
      s[i] += tmp;
    }
    if (t < DH) {
      float* zb = zw + (bh * NC + c) * DH + t;
      float tmp = *zb;
      *zb = sz;
      sz += tmp;
    }
  }
}

// ---------------------------------------------------------------------------
// Kernel 4: per-chunk attention.
//   A   = causal_mask(Q_c K_c^T)                       (f32 acc -> f16)
//   num = A @ V_c + Q_c @ S_prev
//   den = rowsum(A) + Q_c . z_prev   (clamped at 1e-6)
//   O   = num / den  -> f16 [B,T,H*dv]
// grid = (NC, B*H), block = 256.
// ---------------------------------------------------------------------------
__global__ void chunk_attn_kernel(const _Float16* __restrict__ Qh,
                                  const _Float16* __restrict__ Kh,
                                  const _Float16* __restrict__ Vh,
                                  const float* __restrict__ KVw,   // exclusive
                                  const float* __restrict__ zw,    // exclusive
                                  _Float16* __restrict__ Ow) {
  __shared__ __align__(16) _Float16 Qs[64 * 64];
  __shared__ __align__(16) _Float16 Ks[64 * 64];
  __shared__ __align__(16) _Float16 Vs[64 * 64];
  __shared__ __align__(16) _Float16 Ss[64 * 64];   // S_prev (f32->f16)
  __shared__ __align__(16) _Float16 As[64 * 64];   // masked attention weights
  __shared__ float zs[64];
  __shared__ float dinv[64];

  const int c  = blockIdx.x;
  const int bh = blockIdx.y;
  const int b  = bh >> 3, h = bh & 7;
  const int t  = threadIdx.x;
  const int wave = t >> 5, lane = t & 31;
  const int ml = lane & 15, nl = lane & 15, hi8 = (lane >> 4) << 3;

  const long base = (long)(bh * TT + c * CH) * DH;
  for (int r = 0; r < 2; ++r) {
    int idx = t + r * 256;
    ((uint4*)Qs)[idx] = ((const uint4*)(Qh + base))[idx];
    ((uint4*)Ks)[idx] = ((const uint4*)(Kh + base))[idx];
    ((uint4*)Vs)[idx] = ((const uint4*)(Vh + base))[idx];
  }
  const float* Sp = KVw + (long)(bh * NC + c) * (DH * DH);
  for (int i = 0; i < 16; ++i) {
    int e = i * 256 + t;
    Ss[e] = (_Float16)Sp[e];
  }
  if (t < DH) zs[t] = zw[(bh * NC + c) * DH + t];
  __syncthreads();

  // ---- Phase 1: A = mask(Q K^T) ------------------------------------------
  for (int tt = 0; tt < 2; ++tt) {
    int tile = wave * 2 + tt;
    int tm = tile >> 2, tn = tile & 3;               // tm: query rows, tn: key cols
    if (tn > tm) {                                   // fully above diagonal
      for (int v = 0; v < 8; ++v)
        As[(tm * 16 + v + hi8) * 64 + tn * 16 + nl] = (_Float16)0.f;
      continue;
    }
    v8f acc = v8f{};
    for (int ks = 0; ks < 64; ks += 32) {            // contract over dh
      v16h a, bfr;
      for (int i = 0; i < 16; ++i) {
        a[i]   = Qs[(tm * 16 + ml) * 64 + ks + a_k(i, lane)];       // Q[i][d]
        bfr[i] = Ks[(tn * 16 + nl) * 64 + ks + b_k(i, lane)];       // K^T[d][j]
      }
      acc = wmma16(a, bfr, acc);
    }
    for (int v = 0; v < 8; ++v) {
      int m = tm * 16 + v + hi8;
      int n = tn * 16 + nl;
      float val = (n <= m) ? acc[v] : 0.f;           // causal mask (diag kept)
      As[m * 64 + n] = (_Float16)val;
    }
  }
  __syncthreads();

  // ---- den = rowsum(A) + Q . z_prev --------------------------------------
  if (t < CH) {
    float rs = 0.f;
    for (int j = 0; j < CH; ++j) rs += (float)As[t * 64 + j];
    float qz = 0.f;
    for (int d = 0; d < DH; ++d) qz += (float)Qs[t * 64 + d] * zs[d];
    dinv[t] = 1.f / fmaxf(rs + qz, 1e-6f);
  }
  __syncthreads();

  // ---- Phase 2: num = A@V + Q@S_prev ; write O = num * dinv --------------
  for (int tt = 0; tt < 2; ++tt) {
    int tile = wave * 2 + tt;
    int tm = tile >> 2, tn = tile & 3;               // tm: rows, tn: dv cols
    v8f acc = v8f{};
    for (int ks = 0; ks < 64; ks += 32) {            // A @ V (contract j)
      v16h a, bfr;
      for (int i = 0; i < 16; ++i) {
        a[i]   = As[(tm * 16 + ml) * 64 + ks + a_k(i, lane)];
        bfr[i] = Vs[(ks + b_k(i, lane)) * 64 + tn * 16 + nl];
      }
      acc = wmma16(a, bfr, acc);
    }
    for (int ks = 0; ks < 64; ks += 32) {            // Q @ S_prev (contract d)
      v16h a, bfr;
      for (int i = 0; i < 16; ++i) {
        a[i]   = Qs[(tm * 16 + ml) * 64 + ks + a_k(i, lane)];
        bfr[i] = Ss[(ks + b_k(i, lane)) * 64 + tn * 16 + nl];
      }
      acc = wmma16(a, bfr, acc);
    }
    for (int v = 0; v < 8; ++v) {
      int i = tm * 16 + v + hi8;                     // row in chunk
      int j = tn * 16 + nl;                          // dv index
      float val = acc[v] * dinv[i];
      int tok = c * CH + i;
      Ow[((long)(b * TT + tok)) * DM + h * DH + j] = (_Float16)val;
    }
  }
}

// ---------------------------------------------------------------------------
// Kernel 5: output projection  out = O @ W_out^T  (f16 activations, f32 out).
// grid = (8 o-tiles, 64 n-tiles), block = 256.
// ---------------------------------------------------------------------------
__global__ void out_proj_kernel(const _Float16* __restrict__ Ow,
                                const float* __restrict__ Wout,
                                float* __restrict__ out) {
  __shared__ __align__(16) _Float16 Os[64 * 64];
  __shared__ __align__(16) _Float16 Wsh[64 * 64];

  const int o0 = blockIdx.x * 64;
  const int n0 = blockIdx.y * 64;
  const int t  = threadIdx.x;
  const int wave = t >> 5, lane = t & 31;
  const int ml = lane & 15, nl = lane & 15, hi8 = (lane >> 4) << 3;

  v8f acc[2] = {v8f{}, v8f{}};

  for (int k0 = 0; k0 < DM; k0 += 64) {
    for (int r = 0; r < 2; ++r) {                    // stage O tile (f16 direct)
      int idx = t + r * 256;                         // 512 x uint4
      int row = idx >> 3;
      int c8  = (idx & 7) << 3;
      ((uint4*)Os)[idx] =
          *(const uint4*)&Ow[(long)(n0 + row) * DM + k0 + c8];
    }
    for (int r = 0; r < 4; ++r) {                    // stage W_out tile (f32->f16)
      int idx = t + r * 256;
      int row = idx >> 4;
      int c4  = (idx & 15) << 2;
      float4 wv = *(const float4*)&Wout[(o0 + row) * DM + k0 + c4];
      Wsh[row * 64 + c4 + 0] = (_Float16)wv.x;
      Wsh[row * 64 + c4 + 1] = (_Float16)wv.y;
      Wsh[row * 64 + c4 + 2] = (_Float16)wv.z;
      Wsh[row * 64 + c4 + 3] = (_Float16)wv.w;
    }
    __syncthreads();

    for (int tt = 0; tt < 2; ++tt) {
      int tile = wave * 2 + tt;
      int tm = tile >> 2, tn = tile & 3;
      for (int ks = 0; ks < 64; ks += 32) {
        v16h a, b;
        for (int i = 0; i < 16; ++i) {
          a[i] = Os[(tm * 16 + ml) * 64 + ks + a_k(i, lane)];
          b[i] = Wsh[(tn * 16 + nl) * 64 + ks + b_k(i, lane)];
        }
        acc[tt] = wmma16(a, b, acc[tt]);
      }
    }
    __syncthreads();
  }

  for (int tt = 0; tt < 2; ++tt) {
    int tile = wave * 2 + tt;
    int tm = tile >> 2, tn = tile & 3;
    for (int v = 0; v < 8; ++v) {
      int n = n0 + tm * 16 + v + hi8;
      int o = o0 + tn * 16 + nl;
      out[(long)n * DM + o] = acc[tt][v];
    }
  }
}

// ---------------------------------------------------------------------------
extern "C" void kernel_launch(void* const* d_in, const int* in_sizes, int n_in,
                              void* d_out, int out_size, void* d_ws, size_t ws_size,
                              hipStream_t stream) {
  const float* x    = (const float*)d_in[0];
  const float* Wq   = (const float*)d_in[1];
  const float* Wk   = (const float*)d_in[2];
  const float* Wv   = (const float*)d_in[3];
  const float* Wout = (const float*)d_in[4];
  float* out = (float*)d_out;

  char* ws = (char*)d_ws;
  const size_t MB = 1024 * 1024;
  _Float16* Qh = (_Float16*)(ws + 0 * MB);   // [B,H,T,dh] f16 : 4 MB
  _Float16* Kh = (_Float16*)(ws + 4 * MB);   // 4 MB
  _Float16* Vh = (_Float16*)(ws + 8 * MB);   // 4 MB
  _Float16* Ow = (_Float16*)(ws + 12 * MB);  // [B,T,H*dv] f16 : 4 MB
  float*   KVw = (float*)(ws + 16 * MB);     // [B*H,NC,dh,dv] f32 : 8 MB
  float*   zwp = (float*)(ws + 24 * MB);     // [B*H,NC,dh]    f32 : 128 KB

  proj_qkv_kernel<<<dim3(8, 64, 3), 256, 0, stream>>>(x, Wq, Wk, Wv, Qh, Kh, Vh);
  chunk_kv_kernel<<<dim3(NC, BQ * HH), 256, 0, stream>>>(Kh, Vh, KVw, zwp);
  chunk_scan_kernel<<<BQ * HH, 256, 0, stream>>>(KVw, zwp);
  chunk_attn_kernel<<<dim3(NC, BQ * HH), 256, 0, stream>>>(Qh, Kh, Vh, KVw, zwp, Ow);
  out_proj_kernel<<<dim3(8, 64), 256, 0, stream>>>(Ow, Wout, out);
}
